// SpConGAT_63513976373537
// MI455X (gfx1250) — matrix operand order repaired and checked by reading, not compiled, not run on hardware
//
#include <hip/hip_runtime.h>
#include <hip/hip_bf16.h>
#include <math.h>

// Problem constants (match reference)
#define N_NODES 4096
#define NFEAT   512
#define NHID    64
#define NCLASS  16
#define NHEADS  8
#define LRELU_A 0.2f
#define KNN_WAVES 8   // waves per KNN workgroup (column-loop split)

typedef __attribute__((ext_vector_type(16))) _Float16 v16h;
typedef __attribute__((ext_vector_type(2)))  _Float16 v2h;
typedef __attribute__((ext_vector_type(8)))  float    v8f;

__device__ __forceinline__ v8f wmma_f16(v16h a, v16h b, v8f c) {
  // D = A(16x32 f16) * B(32x16 f16) + C(16x16 f32)
  return __builtin_amdgcn_wmma_f32_16x16x32_f16(false, a, false, b, (short)0, c,
                                                false, false);
}

// ---------------------------------------------------------------------------
// k0: convert x (f32) -> f16 and compute row squared norms. One wave per row.
__global__ void __launch_bounds__(32)
cvt_sq_kernel(const float* __restrict__ x1, const float* __restrict__ x2,
              _Float16* __restrict__ xh, float* __restrict__ sq) {
  const int row = blockIdx.x, g = blockIdx.y, l = threadIdx.x;
  const float* x = (g == 0 ? x1 : x2) + (size_t)row * NFEAT;
  _Float16* xo = xh + ((size_t)g * N_NODES + row) * NFEAT;
  float s = 0.f;
  for (int k = l; k < NFEAT; k += 32) { float v = x[k]; s += v * v; xo[k] = (_Float16)v; }
  #pragma unroll
  for (int off = 16; off > 0; off >>= 1) s += __shfl_xor(s, off);
  if (l == 0) sq[g * N_NODES + row] = s;
}

// ---------------------------------------------------------------------------
// k1: convert W [8,512,64] and Wo [512,16] to f16.
__global__ void cvt_w_kernel(const float* __restrict__ W, const float* __restrict__ Wo,
                             _Float16* __restrict__ Wh, _Float16* __restrict__ Woh) {
  const int t = blockIdx.x * blockDim.x + threadIdx.x;
  const int nW = NHEADS * NFEAT * NHID;
  if (t < nW) Wh[t] = (_Float16)W[t];
  else {
    int u = t - nW;
    if (u < NFEAT * NCLASS) Woh[u] = (_Float16)Wo[u];
  }
}

// ---------------------------------------------------------------------------
// k2: KNN via WMMA Gram tiles + row argmin.
// 8 waves per workgroup share one 16-row tile; wave w handles column tiles
// ct = w, w+8, ... (keeps waves on the same B tiles ~simultaneously for L0/L2
// reuse).  Per-wave argmin -> lane butterfly -> LDS cross-wave reduction.
// d2[i,j] ordering uses sq[j] - 2*dot(i,j) (sq[i] is constant per row).
__global__ void __launch_bounds__(32 * KNN_WAVES)
knn_kernel(const _Float16* __restrict__ xh, const float* __restrict__ sq,
           int* __restrict__ nn) {
  const int rt = blockIdx.x, g = blockIdx.y;
  const int w = threadIdx.x >> 5, l = threadIdx.x & 31;
  const int n = l & 15, hi = l >> 4;
  const _Float16* xg = xh + (size_t)g * N_NODES * NFEAT;
  const float* sqg = sq + g * N_NODES;
  const _Float16* arow = xg + (size_t)(rt * 16 + n) * NFEAT;

  __shared__ float svals[KNN_WAVES][16];
  __shared__ int   sidx[KNN_WAVES][16];

  // Hoist A fragments (16 k-steps covering K=512) — reused across all col tiles.
  v16h A[16];
  #pragma unroll
  for (int ks = 0; ks < 16; ++ks) {
    #pragma unroll
    for (int v = 0; v < 8; ++v) {
      const int ka = ks * 32 + hi * 16 + (v << 1);
      v2h p = *(const v2h*)(arow + ka);
      A[ks][2 * v] = p[0]; A[ks][2 * v + 1] = p[1];
    }
  }

  float minv[8]; int mini[8];
  #pragma unroll
  for (int r = 0; r < 8; ++r) { minv[r] = 3.0e38f; mini[r] = 0; }

  for (int ct = w; ct < N_NODES / 16; ct += KNN_WAVES) {
    const _Float16* brow = xg + (size_t)(ct * 16 + n) * NFEAT;
    const float sqc = sqg[ct * 16 + n];
    v8f c = {};
    #pragma unroll
    for (int ks = 0; ks < 16; ++ks) {
      v16h B;
      #pragma unroll
      for (int v = 0; v < 8; ++v) {
        const int kb = ks * 32 + hi * 16 + (v << 1);
        v2h p = *(const v2h*)(brow + kb);
        B[2 * v] = p[0]; B[2 * v + 1] = p[1];
      }
      c = wmma_f16(A[ks], B, c);
    }
    // D layout: lane l holds column n = l%16, rows r + 8*hi in VGPR r.
    #pragma unroll
    for (int r = 0; r < 8; ++r) {
      const int row = rt * 16 + r + 8 * hi;
      const int col = ct * 16 + n;
      const float d = sqc - 2.0f * c[r];
      if (col != row && d < minv[r]) { minv[r] = d; mini[r] = col; }
    }
  }
  // Butterfly argmin over the 16 lanes sharing each row (masks stay in-half).
  #pragma unroll
  for (int r = 0; r < 8; ++r) {
    float mv = minv[r]; int mi = mini[r];
    #pragma unroll
    for (int off = 1; off <= 8; off <<= 1) {
      const float ov = __shfl_xor(mv, off);
      const int   oi = __shfl_xor(mi, off);
      if (ov < mv) { mv = ov; mi = oi; }
    }
    if (n == 0) { svals[w][r + 8 * hi] = mv; sidx[w][r + 8 * hi] = mi; }
  }
  __syncthreads();
  // Cross-wave reduction: first 16 threads each own one row.
  if (threadIdx.x < 16) {
    const int row = threadIdx.x;
    float mv = svals[0][row]; int mi = sidx[0][row];
    #pragma unroll
    for (int ww = 1; ww < KNN_WAVES; ++ww) {
      const float ov = svals[ww][row];
      if (ov < mv) { mv = ov; mi = sidx[ww][row]; }
    }
    nn[g * N_NODES + rt * 16 + row] = mi;
  }
}

// ---------------------------------------------------------------------------
// k3: h[head] = x @ W[head]  (4096x512 @ 512x64), WMMA, one wave per 16x16 tile.
__global__ void __launch_bounds__(32)
gemm_h_kernel(const _Float16* __restrict__ xh, const _Float16* __restrict__ Wh,
              float* __restrict__ h1) {
  const int mt = blockIdx.x;
  const int head = blockIdx.y >> 2, nt = blockIdx.y & 3;
  const int g = blockIdx.z;
  const int l = threadIdx.x, n = l & 15, hi = l >> 4;
  const _Float16* arow = xh + ((size_t)g * N_NODES + mt * 16 + n) * NFEAT;
  const _Float16* wcol = Wh + (size_t)head * NFEAT * NHID + nt * 16 + n;
  v8f c = {};
  for (int ks = 0; ks < 16; ++ks) {
    v16h A, B;
    #pragma unroll
    for (int v = 0; v < 8; ++v) {
      const int ka = ks * 32 + hi * 16 + (v << 1);
      v2h p = *(const v2h*)(arow + ka);
      A[2 * v] = p[0]; A[2 * v + 1] = p[1];
      const int kb = ks * 32 + hi * 16 + (v << 1);
      B[2 * v]     = wcol[(size_t)kb * NHID];
      B[2 * v + 1] = wcol[(size_t)(kb + 1) * NHID];
    }
    c = wmma_f16(A, B, c);
  }
  float* hb = h1 + ((size_t)g * NHEADS + head) * N_NODES * NHID;
  #pragma unroll
  for (int r = 0; r < 8; ++r)
    hb[(size_t)(mt * 16 + r + 8 * hi) * NHID + nt * 16 + n] = c[r];
}

// ---------------------------------------------------------------------------
// k4: per-head attention logit coefficients f = h@a[:64], g = h@a[64:].
__global__ void __launch_bounds__(32)
fg1_kernel(const float* __restrict__ h1, const float* __restrict__ a,
           float* __restrict__ f1, float* __restrict__ g1c) {
  const int row = blockIdx.x, head = blockIdx.y, g = blockIdx.z, l = threadIdx.x;
  const float* hr = h1 + (((size_t)g * NHEADS + head) * N_NODES + row) * NHID;
  const float* av = a + head * 2 * NHID;
  float f  = hr[l] * av[l]         + hr[l + 32] * av[l + 32];
  float gg = hr[l] * av[NHID + l]  + hr[l + 32] * av[NHID + l + 32];
  #pragma unroll
  for (int off = 16; off > 0; off >>= 1) { f += __shfl_xor(f, off); gg += __shfl_xor(gg, off); }
  if (l == 0) {
    const size_t o = ((size_t)g * NHEADS + head) * N_NODES + row;
    f1[o] = f; g1c[o] = gg;
  }
}

// ---------------------------------------------------------------------------
// k5: layer-1 attention as a 2-point gather (mask row = {i, nn(i)}), elu, -> f16 hcat.
__global__ void att1_kernel(const float* __restrict__ h1, const float* __restrict__ f1,
                            const float* __restrict__ g1c, const int* __restrict__ nn,
                            _Float16* __restrict__ hcat) {
  const int row = blockIdx.x, g = blockIdx.y, t = threadIdx.x;
  const int head = t >> 6, d = t & 63;
  const int j = nn[g * N_NODES + row];
  const size_t base = ((size_t)g * NHEADS + head) * N_NODES;
  const float fi = f1[base + row];
  const float s0 = fi + g1c[base + row];
  const float s1 = fi + g1c[base + j];
  const float e0 = expf(-(s0 >= 0.f ? s0 : LRELU_A * s0));
  const float e1 = expf(-(s1 >= 0.f ? s1 : LRELU_A * s1));
  const float hv = h1[(base + row) * NHID + d];
  const float hj = h1[(base + j) * NHID + d];
  float o = (e0 * hv + e1 * hj) / (e0 + e1);
  o = o > 0.f ? o : (expf(o) - 1.f);  // elu
  hcat[((size_t)g * N_NODES + row) * (NHEADS * NHID) + head * NHID + d] = (_Float16)o;
}

// ---------------------------------------------------------------------------
// k6: z = hcat @ Wo  (4096x512 @ 512x16), WMMA, one wave per 16-row tile.
__global__ void __launch_bounds__(32)
gemm_z_kernel(const _Float16* __restrict__ hcat, const _Float16* __restrict__ Woh,
              float* __restrict__ z) {
  const int mt = blockIdx.x, g = blockIdx.y;
  const int l = threadIdx.x, n = l & 15, hi = l >> 4;
  const _Float16* arow = hcat + ((size_t)g * N_NODES + mt * 16 + n) * (NHEADS * NHID);
  v8f c = {};
  for (int ks = 0; ks < 16; ++ks) {
    v16h A, B;
    #pragma unroll
    for (int v = 0; v < 8; ++v) {
      const int ka = ks * 32 + hi * 16 + (v << 1);
      v2h p = *(const v2h*)(arow + ka);
      A[2 * v] = p[0]; A[2 * v + 1] = p[1];
      const int kb = ks * 32 + hi * 16 + (v << 1);
      B[2 * v]     = Woh[(size_t)kb * NCLASS + n];
      B[2 * v + 1] = Woh[(size_t)(kb + 1) * NCLASS + n];
    }
    c = wmma_f16(A, B, c);
  }
  #pragma unroll
  for (int r = 0; r < 8; ++r)
    z[((size_t)g * N_NODES + mt * 16 + r + 8 * hi) * NCLASS + n] = c[r];
}

// ---------------------------------------------------------------------------
// k7: layer-2 logit coefficients (16-dim dot, thread per row).
__global__ void fg2_kernel(const float* __restrict__ z, const float* __restrict__ ao,
                           float* __restrict__ f2, float* __restrict__ g2c) {
  const int row = blockIdx.x * blockDim.x + threadIdx.x, g = blockIdx.y;
  const float* zr = z + ((size_t)g * N_NODES + row) * NCLASS;
  float f = 0.f, gg = 0.f;
  #pragma unroll
  for (int c = 0; c < NCLASS; ++c) { f += zr[c] * ao[c]; gg += zr[c] * ao[NCLASS + c]; }
  f2[g * N_NODES + row] = f; g2c[g * N_NODES + row] = gg;
}

// ---------------------------------------------------------------------------
// k8: layer-2 attention gather + elu.
__global__ void att2_kernel(const float* __restrict__ z, const float* __restrict__ f2,
                            const float* __restrict__ g2c, const int* __restrict__ nn,
                            float* __restrict__ gout) {
  const int t = threadIdx.x;
  const int row = blockIdx.x * 16 + (t >> 4), c = t & 15, g = blockIdx.y;
  const int j = nn[g * N_NODES + row];
  const float fi = f2[g * N_NODES + row];
  const float s0 = fi + g2c[g * N_NODES + row];
  const float s1 = fi + g2c[g * N_NODES + j];
  const float e0 = expf(-(s0 >= 0.f ? s0 : LRELU_A * s0));
  const float e1 = expf(-(s1 >= 0.f ? s1 : LRELU_A * s1));
  const float zi = z[((size_t)g * N_NODES + row) * NCLASS + c];
  const float zj = z[((size_t)g * N_NODES + j) * NCLASS + c];
  float o = (e0 * zi + e1 * zj) / (e0 + e1);
  o = o > 0.f ? o : (expf(o) - 1.f);
  gout[((size_t)g * N_NODES + row) * NCLASS + c] = o;
}

// ---------------------------------------------------------------------------
// k9: out[i] = -( sum_c |g1-g2| * w_mlp1[c] + b ).
__global__ void final_kernel(const float* __restrict__ gout, const float* __restrict__ w,
                             const float* __restrict__ b, float* __restrict__ out) {
  const int row = blockIdx.x * blockDim.x + threadIdx.x;
  const float* p1 = gout + (size_t)row * NCLASS;
  const float* p2 = gout + ((size_t)N_NODES + row) * NCLASS;
  float s = 0.f;
  #pragma unroll
  for (int c = 0; c < NCLASS; ++c) s += fabsf(p1[c] - p2[c]) * w[c];
  out[row] = -(s + b[0]);
}

// ---------------------------------------------------------------------------
extern "C" void kernel_launch(void* const* d_in, const int* in_sizes, int n_in,
                              void* d_out, int out_size, void* d_ws, size_t ws_size,
                              hipStream_t stream) {
  (void)in_sizes; (void)n_in; (void)out_size; (void)ws_size;
  const float* x1 = (const float*)d_in[0];
  const float* x2 = (const float*)d_in[1];
  const float* W  = (const float*)d_in[2];
  const float* a  = (const float*)d_in[3];
  const float* Wo = (const float*)d_in[4];
  const float* ao = (const float*)d_in[5];
  const float* wm = (const float*)d_in[6];
  const float* bm = (const float*)d_in[7];
  float* out = (float*)d_out;

  // Workspace carve-out (~36 MB total), 256B-aligned slices.
  char* p = (char*)d_ws;
  auto carve = [&](size_t bytes) { char* r = p; p += (bytes + 255) & ~(size_t)255; return r; };
  _Float16* xh   = (_Float16*)carve(2ull * N_NODES * NFEAT * sizeof(_Float16));
  _Float16* Wh   = (_Float16*)carve((size_t)NHEADS * NFEAT * NHID * sizeof(_Float16));
  _Float16* Woh  = (_Float16*)carve((size_t)NFEAT * NCLASS * sizeof(_Float16));
  float*    sq   = (float*)   carve(2ull * N_NODES * sizeof(float));
  int*      nn   = (int*)     carve(2ull * N_NODES * sizeof(int));
  float*    h1   = (float*)   carve(2ull * NHEADS * N_NODES * NHID * sizeof(float));
  float*    f1   = (float*)   carve(2ull * NHEADS * N_NODES * sizeof(float));
  float*    g1c  = (float*)   carve(2ull * NHEADS * N_NODES * sizeof(float));
  _Float16* hcat = (_Float16*)carve(2ull * N_NODES * NHEADS * NHID * sizeof(_Float16));
  float*    z    = (float*)   carve(2ull * N_NODES * NCLASS * sizeof(float));
  float*    f2   = (float*)   carve(2ull * N_NODES * sizeof(float));
  float*    g2c  = (float*)   carve(2ull * N_NODES * sizeof(float));
  float*    gout = (float*)   carve(2ull * N_NODES * NCLASS * sizeof(float));

  const dim3 b32(32);
  cvt_sq_kernel<<<dim3(N_NODES, 2), b32, 0, stream>>>(x1, x2, xh, sq);
  const int ncvt = NHEADS * NFEAT * NHID + NFEAT * NCLASS;
  cvt_w_kernel<<<(ncvt + 255) / 256, 256, 0, stream>>>(W, Wo, Wh, Woh);
  knn_kernel<<<dim3(N_NODES / 16, 2), dim3(32 * KNN_WAVES), 0, stream>>>(xh, sq, nn);
  gemm_h_kernel<<<dim3(N_NODES / 16, NHEADS * 4, 2), b32, 0, stream>>>(xh, Wh, h1);
  fg1_kernel<<<dim3(N_NODES, NHEADS, 2), b32, 0, stream>>>(h1, a, f1, g1c);
  att1_kernel<<<dim3(N_NODES, 2), 512, 0, stream>>>(h1, f1, g1c, nn, hcat);
  gemm_z_kernel<<<dim3(N_NODES / 16, 2), b32, 0, stream>>>(hcat, Woh, z);
  fg2_kernel<<<dim3(N_NODES / 64, 2), 64, 0, stream>>>(z, ao, f2, g2c);
  att2_kernel<<<dim3(N_NODES / 16, 2), 256, 0, stream>>>(z, f2, g2c, nn, gout);
  final_kernel<<<dim3(N_NODES / 256), 256, 0, stream>>>(gout, wm, bm, out);
}
